// RFCOS_EARL_70325794505057
// MI455X (gfx1250) — matrix operand order repaired and checked by reading, not compiled, not run on hardware
//
#include <hip/hip_runtime.h>
#include <hip/hip_bf16.h>
#include <stdint.h>

// ---------------------------------------------------------------------------
// Types for WMMA bf16 path (gfx1250, wave32)
// ---------------------------------------------------------------------------
typedef __attribute__((ext_vector_type(16))) __bf16 v16bf;
typedef __attribute__((ext_vector_type(8)))  float  v8f;

struct __attribute__((aligned(16))) U4 { unsigned int x, y, z, w; };

union FragAB {
    v16bf v;
    U4    q[2];
};

static __device__ __forceinline__ unsigned short f32_to_bf16(float f) {
    unsigned int u = __float_as_uint(f);
    // round-to-nearest-even
    unsigned int r = (u + 0x7FFFu + ((u >> 16) & 1u)) >> 16;
    return (unsigned short)r;
}

// Async global->LDS copy of 16 bytes (ASYNCcnt-tracked, gfx1250).
static __device__ __forceinline__ void async_ld_b128(unsigned lds_off,
                                                     const void* gptr) {
    unsigned long long ga = (unsigned long long)(uintptr_t)gptr;
    asm volatile("global_load_async_to_lds_b128 %0, %1, off"
                 :: "v"(lds_off), "v"(ga) : "memory");
}

static __device__ __forceinline__ void wait_async0() {
    asm volatile("s_wait_asynccnt 0" ::: "memory");
}

// ---------------------------------------------------------------------------
// Weight repack: f32 (Cout, 256, 3, 3) -> bf16 [tap][coP][256], zero-padded co
// ---------------------------------------------------------------------------
__global__ void repack_weights(const float* __restrict__ w,
                               unsigned short* __restrict__ out,
                               int Cout, int CoutP) {
    int tid = blockIdx.x * blockDim.x + threadIdx.x;
    int total = 9 * CoutP * 256;
    if (tid >= total) return;
    int tap = tid / (CoutP * 256);
    int rem = tid - tap * (CoutP * 256);
    int co  = rem >> 8;
    int ci  = rem & 255;
    float val = 0.0f;
    if (co < Cout) val = w[(size_t)(co * 256 + ci) * 9 + tap];
    out[(size_t)(tap * CoutP + co) * 256 + ci] = f32_to_bf16(val);
}

__global__ void pad_bias(const float* __restrict__ b, float* __restrict__ out,
                         int Cout, int CoutP) {
    int tid = blockIdx.x * blockDim.x + threadIdx.x;
    if (tid < CoutP) out[tid] = (tid < Cout) ? b[tid] : 0.0f;
}

// ---------------------------------------------------------------------------
// NCHW f32 -> NHWC bf16
// ---------------------------------------------------------------------------
__global__ void nchw_to_nhwc_bf16(const float* __restrict__ in,
                                  unsigned short* __restrict__ out,
                                  int HW) {
    int tid = blockIdx.x * blockDim.x + threadIdx.x;
    int total = 2 * 256 * HW;
    if (tid >= total) return;
    int n = tid / (256 * HW);
    int r = tid - n * (256 * HW);
    int c = r / HW;
    int p = r - c * HW;
    out[(size_t)(n * HW + p) * 256 + c] = f32_to_bf16(in[tid]);
}

// ---------------------------------------------------------------------------
// Implicit-GEMM 3x3 conv, Cin=256, via v_wmma_f32_16x16x32_bf16.
//   Grid: (Mtot/64, CoutP/(16*NSUB)), block 128 (4 waves).
//   Wave w handles 16 pixels (pixel_base = bx*64 + w*16) x (NSUB*16) couts.
//   The 64co x 256ci weight slab for tap t+1 is async-copied (ASYNCcnt) into
//   the alternate LDS buffer while tap t computes; one s_wait_asynccnt +
//   barrier per tap boundary.
// ---------------------------------------------------------------------------
template <int NSUB, bool RELU, bool OUTF32>
__global__ __launch_bounds__(128) void conv3x3_wmma(
    const unsigned short* __restrict__ in,    // bf16 NHWC [Mtot][256]
    const unsigned short* __restrict__ wgt,   // bf16 [9][CoutP][256]
    const float* __restrict__ bias,           // f32  [CoutP]
    unsigned short* __restrict__ out_bf,      // bf16 NHWC [Mtot][CoutP]
    float* __restrict__ out_f,                // f32  NHWC [Mtot][CoutP]
    int H, int W, int HW, int coutP) {

    constexpr int SLAB_U4 = NSUB * 512;             // U4 units per slab
    __shared__ U4 smemW[2 * SLAB_U4];               // double-buffered

    const int lane = threadIdx.x & 31;
    const int wave = threadIdx.x >> 5;
    const int mrow = lane & 15;
    const bool hiK = lane >= 16;

    const int pixel_base = blockIdx.x * 64 + wave * 16;
    const int cout_base  = blockIdx.y * (NSUB * 16);

    // decode this lane's output pixel
    const int p   = pixel_base + mrow;
    const int n   = p / HW;
    const int rem = p - n * HW;
    const int y   = rem / W;
    const int x   = rem - y * W;

    v8f acc[NSUB];
#pragma unroll
    for (int s = 0; s < NSUB; ++s)
#pragma unroll
        for (int i = 0; i < 8; ++i) acc[s][i] = 0.0f;

    // async-stage the weight slab for `tap` into buffer `buf`
    auto stage = [&](int tap, int buf) {
        const unsigned short* wb = wgt + (size_t)(tap * coutP + cout_base) * 256;
        const unsigned base_off = (unsigned)(buf * SLAB_U4 * 16);
#pragma unroll
        for (int i = 0; i < NSUB * 4; ++i) {
            const int u   = threadIdx.x + i * 128;
            const int row = u >> 5;              // 32 x 16B units per 512B row
            const int seg = u & 31;
            async_ld_b128(base_off + (unsigned)u * 16,
                          wb + (size_t)row * 256 + seg * 8);
        }
    };

    stage(0, 0);
    wait_async0();
    __syncthreads();

    for (int tap = 0; tap < 9; ++tap) {
        const int dy = tap / 3 - 1;
        const int dx = tap - (tap / 3) * 3 - 1;
        const int yy = y + dy;
        const int xx = x + dx;
        const bool valid = (yy >= 0) && (yy < H) && (xx >= 0) && (xx < W);
        const unsigned short* abase = in + (size_t)(n * HW + yy * W + xx) * 256;
        if (valid) __builtin_prefetch((const void*)abase, 0, 0);

        // kick off next tap's slab into the alternate buffer
        if (tap < 8) stage(tap + 1, (tap + 1) & 1);

        const unsigned short* sw =
            (const unsigned short*)(smemW + (tap & 1) * SLAB_U4);

#pragma unroll
        for (int kc = 0; kc < 8; ++kc) {
            const int k0 = kc * 32;

            // A fragment: 16-bit A 16x32 layout
            // lanes 0-15: K = {k0..k0+7, k0+16..k0+23}
            // lanes16-31: K = {k0+8..k0+15, k0+24..k0+31}
            FragAB a;
            if (valid) {
                a.q[0] = *(const U4*)(abase + k0 + (hiK ? 8 : 0));
                a.q[1] = *(const U4*)(abase + k0 + 16 + (hiK ? 8 : 0));
            } else {
                U4 z; z.x = 0u; z.y = 0u; z.z = 0u; z.w = 0u;
                a.q[0] = z; a.q[1] = z;
            }

            // batch all B-fragment LDS reads, then issue the WMMAs
            FragAB b[NSUB];
#pragma unroll
            for (int s = 0; s < NSUB; ++s) {
                const U4* pb = (const U4*)(sw + (s * 16 + mrow) * 256 + k0 +
                                           (hiK ? 8 : 0));
                b[s].q[0] = pb[0];
                b[s].q[1] = pb[2];
            }
#pragma unroll
            for (int s = 0; s < NSUB; ++s) {
                acc[s] = __builtin_amdgcn_wmma_f32_16x16x32_bf16(
                    false, a.v, false, b[s].v, (short)0, acc[s], false, false);
            }
        }

        wait_async0();     // next slab landed (this wave's share)
        __syncthreads();   // all waves' shares landed / slab consumed
    }

    // Epilogue: D layout — VGPR r holds (M = r + 8*hiK, N = lane&15)
#pragma unroll
    for (int s = 0; s < NSUB; ++s) {
        const int co = cout_base + s * 16 + mrow;
        const float bv = bias[co];
#pragma unroll
        for (int r = 0; r < 8; ++r) {
            const int pp = pixel_base + r + (hiK ? 8 : 0);
            float v = acc[s][r] + bv;
            if (RELU) v = v > 0.0f ? v : 0.0f;
            if (OUTF32) out_f[(size_t)pp * coutP + co] = v;
            else        out_bf[(size_t)pp * coutP + co] = f32_to_bf16(v);
        }
    }
}

// ---------------------------------------------------------------------------
// sigmoid + threshold: logits f32 [2*HW][16] (15 valid) -> sc [2][HW*15]
// ---------------------------------------------------------------------------
__global__ void score_sigmoid(const float* __restrict__ logits,
                              float* __restrict__ sc, int HW) {
    int tid = blockIdx.x * blockDim.x + threadIdx.x;
    int total = 2 * HW * 15;
    if (tid >= total) return;
    int n = tid / (HW * 15);
    int r = tid - n * (HW * 15);
    int p = r / 15;
    int c = r - p * 15;
    float l = logits[(size_t)(n * HW + p) * 16 + c];
    float s = 1.0f / (1.0f + __expf(-l));
    sc[(size_t)n * (HW * 15) + r] = (s > 0.05f) ? s : 0.0f;
}

// ---------------------------------------------------------------------------
// Exact top-k (descending value, ascending index on ties, matching lax.top_k)
// One block per image; iterative max-selection; gathers boxes and writes
// out[n][lvl_off + i][0..5] = {score, reg[0..4]}.
// sc is fully rewritten by score_sigmoid each launch -> deterministic.
// ---------------------------------------------------------------------------
__global__ __launch_bounds__(1024) void topk_select(
    float* sc,                         // [2][len], mutated
    const float* __restrict__ reg,     // f32 [2*HW][16] (5 valid)
    float* __restrict__ out,           // [2][total_k][6]
    int HW, int len, int k, int lvl_off, int total_k) {

    __shared__ float svals[1024];
    __shared__ int   sidx[1024];

    const int n   = blockIdx.x;
    const int tid = threadIdx.x;
    float* scn = sc + (size_t)n * len;

    for (int r = 0; r < k; ++r) {
        float best = -3.0e38f;
        int   bi   = len;
        for (int i = tid; i < len; i += 1024) {
            float v = scn[i];
            if (v > best) { best = v; bi = i; }
        }
        svals[tid] = best;
        sidx[tid]  = bi;
        __syncthreads();
        for (int s = 512; s > 0; s >>= 1) {
            if (tid < s) {
                float v2 = svals[tid + s];
                int   i2 = sidx[tid + s];
                if (v2 > svals[tid] || (v2 == svals[tid] && i2 < sidx[tid])) {
                    svals[tid] = v2;
                    sidx[tid]  = i2;
                }
            }
            __syncthreads();
        }
        if (tid == 0) {
            int   idx = sidx[0];
            float val = svals[0];
            int prow  = idx / 15;
            float* o  = out + ((size_t)n * total_k + lvl_off + r) * 6;
            const float* rg = reg + (size_t)(n * HW + prow) * 16;
            o[0] = val;
            o[1] = rg[0]; o[2] = rg[1]; o[3] = rg[2]; o[4] = rg[3]; o[5] = rg[4];
            scn[idx] = -1.0f;
        }
        __syncthreads();
    }
}

// ---------------------------------------------------------------------------
// Host launcher
// ---------------------------------------------------------------------------
extern "C" void kernel_launch(void* const* d_in, const int* in_sizes, int n_in,
                              void* d_out, int out_size, void* d_ws, size_t ws_size,
                              hipStream_t stream) {
    (void)in_sizes; (void)n_in; (void)out_size; (void)ws_size;

    const float* feat[5];
    for (int i = 0; i < 5; ++i) feat[i] = (const float*)d_in[i];
    const float *cls_w[4], *cls_b[4], *box_w[4], *box_b[4];
    for (int j = 0; j < 4; ++j) {
        cls_w[j] = (const float*)d_in[5 + 4 * j];
        cls_b[j] = (const float*)d_in[6 + 4 * j];
        box_w[j] = (const float*)d_in[7 + 4 * j];
        box_b[j] = (const float*)d_in[8 + 4 * j];
    }
    const float* score_w = (const float*)d_in[21];
    const float* score_b = (const float*)d_in[22];
    const float* pred_w  = (const float*)d_in[23];
    const float* pred_b  = (const float*)d_in[24];

    // ---- workspace carve-out -------------------------------------------
    char* wsp = (char*)d_ws;
    auto carve = [&](size_t bytes) -> char* {
        char* pc = wsp;
        wsp += (bytes + 255) & ~((size_t)255);
        return pc;
    };
    const size_t wBigB   = (size_t)9 * 256 * 256 * 2;   // bf16
    const size_t wSmallB = (size_t)9 * 16 * 256 * 2;
    unsigned short *wCls[4], *wBox[4];
    for (int j = 0; j < 4; ++j) wCls[j] = (unsigned short*)carve(wBigB);
    for (int j = 0; j < 4; ++j) wBox[j] = (unsigned short*)carve(wBigB);
    unsigned short* wScore = (unsigned short*)carve(wSmallB);
    unsigned short* wPred  = (unsigned short*)carve(wSmallB);
    float *bCls[4], *bBox[4];
    for (int j = 0; j < 4; ++j) bCls[j] = (float*)carve(256 * 4);
    for (int j = 0; j < 4; ++j) bBox[j] = (float*)carve(256 * 4);
    float* bScore = (float*)carve(16 * 4);
    float* bPred  = (float*)carve(16 * 4);

    const size_t actB = (size_t)32768 * 256 * 2;        // max level, bf16 NHWC
    unsigned short* featB = (unsigned short*)carve(actB);
    unsigned short* actA  = (unsigned short*)carve(actB);
    unsigned short* actBf = (unsigned short*)carve(actB);
    float* logits = (float*)carve((size_t)32768 * 16 * 4);
    float* regbuf = (float*)carve((size_t)32768 * 16 * 4);
    float* scores = (float*)carve((size_t)2 * 16384 * 15 * 4);

    // ---- weight / bias repack ------------------------------------------
    {
        int totB = 9 * 256 * 256;
        dim3 g((totB + 255) / 256), b(256);
        for (int j = 0; j < 4; ++j) {
            repack_weights<<<g, b, 0, stream>>>(cls_w[j], wCls[j], 256, 256);
            repack_weights<<<g, b, 0, stream>>>(box_w[j], wBox[j], 256, 256);
        }
        int totS = 9 * 16 * 256;
        dim3 gs((totS + 255) / 256);
        repack_weights<<<gs, b, 0, stream>>>(score_w, wScore, 15, 16);
        repack_weights<<<gs, b, 0, stream>>>(pred_w, wPred, 5, 16);
        for (int j = 0; j < 4; ++j) {
            pad_bias<<<1, 256, 0, stream>>>(cls_b[j], bCls[j], 256, 256);
            pad_bias<<<1, 256, 0, stream>>>(box_b[j], bBox[j], 256, 256);
        }
        pad_bias<<<1, 256, 0, stream>>>(score_b, bScore, 15, 16);
        pad_bias<<<1, 256, 0, stream>>>(pred_b, bPred, 5, 16);
    }

    // ---- per-level pipeline --------------------------------------------
    const int HS[5] = {128, 64, 32, 16, 8};
    int lvl_off = 0;
    const int total_k = 4960;
    float* outp = (float*)d_out;

    for (int l = 0; l < 5; ++l) {
        const int H = HS[l], W = HS[l], HW = H * W, Mtot = 2 * HW;

        {   // NCHW f32 -> NHWC bf16
            int tot = 2 * 256 * HW;
            nchw_to_nhwc_bf16<<<dim3((tot + 255) / 256), dim3(256), 0, stream>>>(
                feat[l], featB, HW);
        }

        // ---- cls branch: 4x (conv+relu) then score predictor -----------
        const unsigned short* src = featB;
        unsigned short* dst = actA;
        for (int j = 0; j < 4; ++j) {
            conv3x3_wmma<4, true, false><<<dim3(Mtot / 64, 4), dim3(128), 0, stream>>>(
                src, wCls[j], bCls[j], dst, nullptr, H, W, HW, 256);
            src = dst;
            dst = (dst == actA) ? actBf : actA;
        }
        conv3x3_wmma<1, false, true><<<dim3(Mtot / 64, 1), dim3(128), 0, stream>>>(
            src, wScore, bScore, nullptr, logits, H, W, HW, 16);
        {
            int tot = 2 * HW * 15;
            score_sigmoid<<<dim3((tot + 255) / 256), dim3(256), 0, stream>>>(
                logits, scores, HW);
        }

        // ---- box branch: 4x (conv+relu) then box predictor -------------
        src = featB;
        dst = actA;
        for (int j = 0; j < 4; ++j) {
            conv3x3_wmma<4, true, false><<<dim3(Mtot / 64, 4), dim3(128), 0, stream>>>(
                src, wBox[j], bBox[j], dst, nullptr, H, W, HW, 256);
            src = dst;
            dst = (dst == actA) ? actBf : actA;
        }
        conv3x3_wmma<1, false, true><<<dim3(Mtot / 64, 1), dim3(128), 0, stream>>>(
            src, wPred, bPred, nullptr, regbuf, H, W, HW, 16);

        // ---- top-k + gather --------------------------------------------
        const int len = HW * 15;
        const int kk  = (len < 1000) ? len : 1000;
        topk_select<<<dim3(2), dim3(1024), 0, stream>>>(
            scores, regbuf, outp, HW, len, kk, lvl_off, total_k);
        lvl_off += kk;
    }
}
